// ContinualLoraMoeOneGateInjectedLinear_3221225472404
// MI455X (gfx1250) — matrix operand order repaired and checked by reading, not compiled, not run on hardware
//
#include <hip/hip_runtime.h>

// Problem constants (match setup_inputs)
#define B_   4
#define S_   4096
#define IN_  1024
#define OUT_ 1024
#define P_   5
#define R_   8
#define NUM_TASKS_ 5

typedef __bf16 v8bf  __attribute__((ext_vector_type(8)));
typedef __bf16 v16bf __attribute__((ext_vector_type(16)));
typedef float  v8f   __attribute__((ext_vector_type(8)));

// ---- bf16 helpers (RNE, bit-exact, no header dependence) -------------------
__device__ __forceinline__ unsigned short f2bf(float f) {
  unsigned int u = __float_as_uint(f);
  u += 0x7fffu + ((u >> 16) & 1u);
  return (unsigned short)(u >> 16);
}
__device__ __forceinline__ float bf2f(unsigned short h) {
  return __uint_as_float(((unsigned int)h) << 16);
}

// ---------------------------------------------------------------------------
// Kernel 1: partial gate computation (softmax over experts, partial S-sums).
// ---------------------------------------------------------------------------
#define GATE_CHUNKS 32
#define GATE_ROWS   (S_ / GATE_CHUNKS)   // 128 rows per block
#define GATE_WAVES  8                    // 256 threads

__global__ __launch_bounds__(256) void gate_partial_kernel(
    const float* __restrict__ x,
    const float* __restrict__ lora_route,   // [P, IN, P]
    const int*   __restrict__ task_id_p,
    float*       __restrict__ partial)      // [B, GATE_CHUNKS, P]
{
  __shared__ float routeS[IN_ * P_];        // 20 KB
  __shared__ float wAcc[GATE_WAVES][P_];

  const int b  = blockIdx.y;
  const int s0 = blockIdx.x * GATE_ROWS;
  const int t  = threadIdx.x;
  const int tk = *task_id_p;

  for (int idx = t; idx < IN_ * P_; idx += blockDim.x) {
    float v;
    if (tk <= NUM_TASKS_) {
      v = lora_route[(size_t)(tk - 1) * IN_ * P_ + idx];
    } else {
      v = 0.f;
      for (int q = 0; q < P_; ++q) v += lora_route[(size_t)q * IN_ * P_ + idx];
    }
    routeS[idx] = v;
  }
  __syncthreads();

  const int lane = t & 31;
  const int wave = t >> 5;

  float wsum[P_];
  #pragma unroll
  for (int p = 0; p < P_; ++p) wsum[p] = 0.f;

  for (int sl = wave; sl < GATE_ROWS; sl += GATE_WAVES) {
    const float* xr = x + ((size_t)b * S_ + (s0 + sl)) * IN_;
    float a[P_];
    #pragma unroll
    for (int p = 0; p < P_; ++p) a[p] = 0.f;
    for (int i = lane; i < IN_; i += 32) {
      float xv = xr[i];
      #pragma unroll
      for (int p = 0; p < P_; ++p) a[p] += xv * routeS[i * P_ + p];
    }
    #pragma unroll
    for (int p = 0; p < P_; ++p) {
      #pragma unroll
      for (int off = 16; off > 0; off >>= 1)
        a[p] += __shfl_xor(a[p], off, 32);
    }
    float m = a[0];
    #pragma unroll
    for (int p = 1; p < P_; ++p) m = fmaxf(m, a[p]);
    float e[P_], s = 0.f;
    #pragma unroll
    for (int p = 0; p < P_; ++p) { e[p] = __expf(a[p] - m); s += e[p]; }
    float inv = 1.f / s;
    #pragma unroll
    for (int p = 0; p < P_; ++p) wsum[p] += e[p] * inv;
  }

  if (lane == 0) {
    #pragma unroll
    for (int p = 0; p < P_; ++p) wAcc[wave][p] = wsum[p];
  }
  __syncthreads();

  if (t == 0) {
    #pragma unroll
    for (int p = 0; p < P_; ++p) {
      float s = 0.f;
      for (int w = 0; w < GATE_WAVES; ++w) s += wAcc[w][p];  // fixed order
      partial[((size_t)b * GATE_CHUNKS + blockIdx.x) * P_ + p] = s;
    }
  }
}

// ---------------------------------------------------------------------------
// Kernel 2: gate[b,p] = 2 * mean_s(omegas) - 1
// ---------------------------------------------------------------------------
__global__ void gate_finalize_kernel(const float* __restrict__ partial,
                                     float* __restrict__ gate) {
  int t = threadIdx.x;
  if (t < B_ * P_) {
    int b = t / P_, p = t % P_;
    float s = 0.f;
    for (int c = 0; c < GATE_CHUNKS; ++c)
      s += partial[((size_t)b * GATE_CHUNKS + c) * P_ + p];
    gate[t] = 2.f * (s * (1.f / (float)S_)) - 1.f;
  }
}

// ---------------------------------------------------------------------------
// Kernel 3: fold per-batch effective weight, directly in GEMM-B layout [o][i]
//   (linear_w is already [o][i] -> coalesced), and split fp32 -> bf16 hi/lo.
//   Wt_{h,l}[b,o,i] ~ linear_w[o,i] + sum_{p<tid} gate[b,p] sum_r down[p,i,r] up[p,r,o]
// ---------------------------------------------------------------------------
__global__ __launch_bounds__(256) void fold_weight_kernel(
    const float* __restrict__ linear_w,    // [OUT, IN]
    const float* __restrict__ lora_down,   // [P, IN, R]
    const float* __restrict__ lora_up,     // [P, R, OUT]
    const float* __restrict__ gate,        // [B, P]
    const int*   __restrict__ task_id_p,
    unsigned short* __restrict__ Wt_h,     // [B, OUT, IN] bf16 hi
    unsigned short* __restrict__ Wt_l)     // [B, OUT, IN] bf16 lo
{
  __shared__ float u[P_ * R_];
  const int o = blockIdx.x;
  const int b = blockIdx.y;
  const int t = threadIdx.x;
  const int tid_eff = min(*task_id_p, NUM_TASKS_);

  if (t < P_ * R_) {
    int p = t / R_, r = t % R_;
    u[t] = (p < tid_eff)
               ? gate[b * P_ + p] * lora_up[((size_t)p * R_ + r) * OUT_ + o]
               : 0.f;
  }
  __syncthreads();

  const size_t dstBase = ((size_t)b * OUT_ + o) * IN_;
  for (int i = t; i < IN_; i += blockDim.x) {
    float w = linear_w[(size_t)o * IN_ + i];
    #pragma unroll
    for (int p = 0; p < P_; ++p)
      #pragma unroll
      for (int r = 0; r < R_; ++r)
        w += u[p * R_ + r] * lora_down[((size_t)p * IN_ + i) * R_ + r];
    unsigned short h = f2bf(w);
    unsigned short l = f2bf(w - bf2f(h));
    Wt_h[dstBase + i] = h;
    Wt_l[dstBase + i] = l;
  }
}

// ---------------------------------------------------------------------------
// Kernel 4: batched GEMM via 3xBF16 split with V_WMMA_F32_16X16X32_BF16
//   out[b] = x[b] @ W_b,  x split hi/lo on the fly, W pre-split in ws.
// Block 256 thr (8 waves), tile 128x128, K-stage 64 (2 wmma K-steps).
// LDS (bf16, row stride 72 -> every fragment load 16B aligned):
//   A{h,l}[m][k]  (k contiguous),  B{h,l}[n][k]  (global layout already [o][i])
// Wave tile 32x64 -> 2x4 fragments; 3 wmma per fragment pair:
//   acc += ah*bh + ah*bl + al*bh      (xl*Wl dropped, ~2^-16 rel. error)
// ---------------------------------------------------------------------------
#define BM  128
#define BN  128
#define BKB 64
#define LDA 72
#define LDB 72

struct BfPair { v8bf lo, hi; };
__device__ __forceinline__ v16bf make_frag(const unsigned short* p0,
                                           const unsigned short* p1) {
  BfPair t;
  t.lo = *(const v8bf*)p0;
  t.hi = *(const v8bf*)p1;
  return __builtin_bit_cast(v16bf, t);
}

__global__ __launch_bounds__(256) void lora_gemm_bf16x3_kernel(
    const float* __restrict__ x,             // [B, S, IN] fp32
    const unsigned short* __restrict__ Wt_h, // [B, OUT, IN] bf16
    const unsigned short* __restrict__ Wt_l, // [B, OUT, IN] bf16
    float* __restrict__ out)                 // [B, S, OUT] fp32
{
  __shared__ unsigned short Ah[BM * LDA];   // 18 KB each
  __shared__ unsigned short Al[BM * LDA];
  __shared__ unsigned short Bh[BN * LDB];
  __shared__ unsigned short Bl[BN * LDB];

  const int b  = blockIdx.z;
  const int n0 = blockIdx.x * BN;
  const int m0 = blockIdx.y * BM;
  const int t    = threadIdx.x;
  const int lane = t & 31;
  const int wave = t >> 5;
  const int half = lane >> 4;
  const int l16  = lane & 15;
  const int waveM = wave & 3;   // 0..3 -> 32-row strip
  const int waveN = wave >> 2;  // 0..1 -> 64-col strip

  const float* xb = x + (size_t)b * S_ * IN_;
  const unsigned short* wh = Wt_h + (size_t)b * OUT_ * IN_;
  const unsigned short* wl = Wt_l + (size_t)b * OUT_ * IN_;

  v8f acc[2][4];
  #pragma unroll
  for (int mi = 0; mi < 2; ++mi)
    #pragma unroll
    for (int ni = 0; ni < 4; ++ni)
      #pragma unroll
      for (int q = 0; q < 8; ++q) acc[mi][ni][q] = 0.f;

  for (int k0 = 0; k0 < IN_; k0 += BKB) {
    // --- stage A: 128 x 64 fp32 tile, split to bf16 hi/lo on the fly ---
    #pragma unroll
    for (int it = 0; it < 8; ++it) {
      int idx = t + it * 256;        // 0..2047
      int row = idx >> 4;            // 0..127
      int q   = idx & 15;            // float4 quads over 64 k
      float4 v = *(const float4*)(xb + (size_t)(m0 + row) * IN_ + k0 + q * 4);
      unsigned short h0 = f2bf(v.x), h1 = f2bf(v.y), h2 = f2bf(v.z), h3 = f2bf(v.w);
      unsigned short l0 = f2bf(v.x - bf2f(h0));
      unsigned short l1 = f2bf(v.y - bf2f(h1));
      unsigned short l2 = f2bf(v.z - bf2f(h2));
      unsigned short l3 = f2bf(v.w - bf2f(h3));
      int base = row * LDA + q * 4;
      *(unsigned int*)&Ah[base]     = (unsigned int)h0 | ((unsigned int)h1 << 16);
      *(unsigned int*)&Ah[base + 2] = (unsigned int)h2 | ((unsigned int)h3 << 16);
      *(unsigned int*)&Al[base]     = (unsigned int)l0 | ((unsigned int)l1 << 16);
      *(unsigned int*)&Al[base + 2] = (unsigned int)l2 | ((unsigned int)l3 << 16);
    }
    // --- stage B: 128 n-rows x 64 k bf16, hi+lo, straight uint4 copies ---
    #pragma unroll
    for (int it = 0; it < 8; ++it) {
      int idx = t + it * 256;        // 0..2047
      int mat = idx >> 10;           // 0 = hi, 1 = lo
      int r   = idx & 1023;
      int nr  = r >> 3;              // 0..127
      int q   = r & 7;               // 8 uint4 per 64-k row
      const unsigned short* src =
          (mat ? wl : wh) + (size_t)(n0 + nr) * IN_ + k0 + q * 8;
      uint4 v = *(const uint4*)src;
      unsigned short* dst = (mat ? Bl : Bh) + nr * LDB + q * 8;
      *(uint4*)dst = v;
    }
    __syncthreads();

    #pragma unroll
    for (int kk = 0; kk < BKB; kk += 32) {
      // A fragments: v16bf elems 0-7 <-> K = kk+8*half.., 8-15 <-> K = kk+16+8*half..
      v16bf ah[2], al[2];
      #pragma unroll
      for (int mi = 0; mi < 2; ++mi) {
        int m = waveM * 32 + mi * 16 + l16;
        const unsigned short* pr = &Ah[m * LDA + kk];
        ah[mi] = make_frag(pr + 8 * half, pr + 16 + 8 * half);
        pr = &Al[m * LDA + kk];
        al[mi] = make_frag(pr + 8 * half, pr + 16 + 8 * half);
      }
      // B fragments: elems 0-7 <-> K = kk+8*half.., 8-15 <-> K = kk+16+8*half.., N = l16
      #pragma unroll
      for (int ni = 0; ni < 4; ++ni) {
        int n = waveN * 64 + ni * 16 + l16;
        v16bf bh = make_frag(&Bh[n * LDB + kk + 8 * half], &Bh[n * LDB + kk + 16 + 8 * half]);
        v16bf bl = make_frag(&Bl[n * LDB + kk + 8 * half], &Bl[n * LDB + kk + 16 + 8 * half]);
        #pragma unroll
        for (int mi = 0; mi < 2; ++mi) {
          acc[mi][ni] = __builtin_amdgcn_wmma_f32_16x16x32_bf16(
              false, ah[mi], false, bh, (short)0, acc[mi][ni], false, false);
          acc[mi][ni] = __builtin_amdgcn_wmma_f32_16x16x32_bf16(
              false, ah[mi], false, bl, (short)0, acc[mi][ni], false, false);
          acc[mi][ni] = __builtin_amdgcn_wmma_f32_16x16x32_bf16(
              false, al[mi], false, bh, (short)0, acc[mi][ni], false, false);
        }
      }
    }
    __syncthreads();
  }

  // --- epilogue: C/D layout VGPR v <-> M = v + 8*half, lane l16 = N ---
  float* ob = out + (size_t)b * S_ * OUT_;
  #pragma unroll
  for (int mi = 0; mi < 2; ++mi) {
    int rowBase = m0 + waveM * 32 + mi * 16 + 8 * half;
    #pragma unroll
    for (int ni = 0; ni < 4; ++ni) {
      int col = n0 + waveN * 64 + ni * 16 + l16;
      #pragma unroll
      for (int v = 0; v < 8; ++v)
        ob[(size_t)(rowBase + v) * OUT_ + col] = acc[mi][ni][v];
    }
  }
}

// ---------------------------------------------------------------------------
// Host launcher
// ---------------------------------------------------------------------------
extern "C" void kernel_launch(void* const* d_in, const int* in_sizes, int n_in,
                              void* d_out, int out_size, void* d_ws, size_t ws_size,
                              hipStream_t stream) {
  const float* x          = (const float*)d_in[0];  // [B,S,IN]
  const float* linear_w   = (const float*)d_in[1];  // [OUT,IN]
  const float* lora_down  = (const float*)d_in[2];  // [P,IN,R]
  const float* lora_up    = (const float*)d_in[3];  // [P,R,OUT]
  const float* lora_route = (const float*)d_in[4];  // [P,IN,P]
  const int*   task_id    = (const int*)d_in[5];    // scalar
  float*       out        = (float*)d_out;          // [B,S,OUT]

  // Workspace layout (~17 MB):
  const size_t wElems = (size_t)B_ * OUT_ * IN_;    // 4.19M
  unsigned short* Wt_h = (unsigned short*)d_ws;     // bf16 hi
  unsigned short* Wt_l = Wt_h + wElems;             // bf16 lo
  float* partial = (float*)(Wt_l + wElems);         // B*CHUNKS*P (16B-aligned)
  float* gate    = partial + (size_t)B_ * GATE_CHUNKS * P_;

  gate_partial_kernel<<<dim3(GATE_CHUNKS, B_), 256, 0, stream>>>(
      x, lora_route, task_id, partial);
  gate_finalize_kernel<<<1, 64, 0, stream>>>(partial, gate);
  fold_weight_kernel<<<dim3(OUT_, B_), 256, 0, stream>>>(
      linear_w, lora_down, lora_up, gate, task_id, Wt_h, Wt_l);
  lora_gemm_bf16x3_kernel<<<dim3(OUT_ / BN, S_ / BM, B_), 256, 0, stream>>>(
      x, Wt_h, Wt_l, out);
}
